// MultiInputLSTMCell_V1_40261023432819
// MI455X (gfx1250) — matrix operand order repaired and checked by reading, not compiled, not run on hardware
//
#include <hip/hip_runtime.h>
#include <math.h>

// ---------------------------------------------------------------------------
// MultiInputLSTMCell on MI455X (gfx1250, wave32, WMMA)
// B=4096, D=512, H=512, X=16
// ---------------------------------------------------------------------------

#define BB 4096
#define DD 512
#define HH 512
#define XX 16
#define LDP 34   // LDS pitch (ushort) to avoid bank conflicts

typedef __attribute__((ext_vector_type(16))) __bf16 v16bf;
typedef __attribute__((ext_vector_type(2)))  __bf16 v2bf;
typedef __attribute__((ext_vector_type(8)))  float  v8f;

union FragB16 {
  v16bf v;
  unsigned u32[8];
  unsigned short u16[16];
};

#if __has_builtin(__builtin_amdgcn_cvt_pk_bf16_f32)
__device__ __forceinline__ unsigned pack2bf(float x, float y) {
  v2bf r = __builtin_amdgcn_cvt_pk_bf16_f32(x, y);
  return __builtin_bit_cast(unsigned, r);
}
#else
// Native __bf16 casts: clang should lower FPTRUNC to hardware bf16 converts
// on gfx1250 (bf16 is a first-class VALU type on CDNA5).
__device__ __forceinline__ unsigned pack2bf(float x, float y) {
  v2bf r;
  r.x = (__bf16)x;
  r.y = (__bf16)y;
  return __builtin_bit_cast(unsigned, r);
}
#endif

// A fragment: 16x32 bf16, lane(0-15)=M, half selects K-halves per ISA 7.12.2
__device__ __forceinline__ v16bf load_frag_a(const unsigned short* ldsA, int mrow, int half) {
  FragB16 f;
#pragma unroll
  for (int v = 0; v < 8; ++v) {
    int kb = ((v < 4) ? (2 * v) : (8 + 2 * v)) + half * 8;  // K pair base
    f.u32[v] = *(const unsigned*)(ldsA + mrow * LDP + kb);
  }
  return f.v;
}

// B fragment: 32x16 bf16, lane(0-15)=N, half selects K-halves (same as A); LDS holds B^T [n][k]
__device__ __forceinline__ v16bf load_frag_b(const unsigned short* ldsB, int ncol, int half) {
  FragB16 f;
#pragma unroll
  for (int v = 0; v < 8; ++v) {
    int kb = ((v < 4) ? (2 * v) : (8 + 2 * v)) + half * 8;
    f.u32[v] = *(const unsigned*)(ldsB + ncol * LDP + kb);
  }
  return f.v;
}

// ---------------------------------------------------------------------------
// Generic bf16 GEMM: C[m,n] = sum_k A(m,k) * B(k,n) + bias[n]
// A(m,k) = k < splitK ? A0[m*512+k] : A1[m*512+k-splitK]  (chunk-invariant)
// B(k,n) = k < splitK ? B0[k*ldb+n] : B1[(k-splitK)*ldb+n]
// Block tile 128x128, 256 threads (8 waves: 4 along M x 2 along N).
// Register-pipelined global->LDS staging, packed bf16 stores.
// ---------------------------------------------------------------------------
__global__ __launch_bounds__(256)
void gemm_bf16_kernel(const float* __restrict__ A0, const float* __restrict__ A1,
                      const float* __restrict__ B0, const float* __restrict__ B1,
                      const float* __restrict__ bias, float* __restrict__ C,
                      int N, int K, int splitK, int ldb) {
  __shared__ unsigned short ldsA[128 * LDP];
  __shared__ unsigned short ldsB[128 * LDP];
  const int tid  = threadIdx.x;
  const int lane = tid & 31;
  const int wave = tid >> 5;
  const int half = lane >> 4;
  const int lan  = lane & 15;
  const int m0 = blockIdx.x * 128;
  const int n0 = blockIdx.y * 128;
  const int wm = (wave & 3) * 32;   // wave M offset (2 m-tiles)
  const int wn = (wave >> 2) * 64;  // wave N offset (4 n-tiles)

  v8f acc[2][4];
#pragma unroll
  for (int a = 0; a < 2; ++a)
#pragma unroll
    for (int b = 0; b < 4; ++b) acc[a][b] = v8f{};

  float2 rA[8];
  float  rB0[8], rB1[8];

  auto loadA = [&](int k0) {
    const float* As = (k0 < splitK) ? (A0 + k0) : (A1 + (k0 - splitK));
#pragma unroll
    for (int e = 0; e < 8; ++e) {
      int i = tid + e * 256;
      int r = i >> 4, c2 = (i & 15) * 2;
      rA[e] = *(const float2*)(As + (size_t)(m0 + r) * 512 + c2);
    }
  };
  auto loadB = [&](int k0) {
    const float* Bs = (k0 < splitK) ? (B0 + (size_t)k0 * ldb)
                                    : (B1 + (size_t)(k0 - splitK) * ldb);
#pragma unroll
    for (int e = 0; e < 8; ++e) {
      int i = tid + e * 256;
      int k2 = (i >> 7) * 2, nc = i & 127;
      rB0[e] = Bs[(size_t)k2 * ldb + n0 + nc];
      rB1[e] = Bs[(size_t)(k2 + 1) * ldb + n0 + nc];
    }
  };

  loadA(0);
  loadB(0);

  for (int k0 = 0; k0 < K; k0 += 32) {
    // convert + store staged registers to LDS (packed bf16 pairs)
#pragma unroll
    for (int e = 0; e < 8; ++e) {
      int i = tid + e * 256;
      int r = i >> 4, c2 = (i & 15) * 2;
      *(unsigned*)(ldsA + r * LDP + c2) = pack2bf(rA[e].x, rA[e].y);
    }
#pragma unroll
    for (int e = 0; e < 8; ++e) {
      int i = tid + e * 256;
      int k2 = (i >> 7) * 2, nc = i & 127;
      *(unsigned*)(ldsB + nc * LDP + k2) = pack2bf(rB0[e], rB1[e]);
    }
    __syncthreads();

    // prefetch next chunk into registers while WMMAs consume LDS
    if (k0 + 32 < K) {
      loadA(k0 + 32);
      loadB(k0 + 32);
    }

    v16bf fa0 = load_frag_a(ldsA, wm + lan, half);
    v16bf fa1 = load_frag_a(ldsA, wm + 16 + lan, half);
#pragma unroll
    for (int tn = 0; tn < 4; ++tn) {
      v16bf fb = load_frag_b(ldsB, wn + tn * 16 + lan, half);
      acc[0][tn] = __builtin_amdgcn_wmma_f32_16x16x32_bf16(false, fa0, false, fb,
                                                           (short)0, acc[0][tn], false, false);
      acc[1][tn] = __builtin_amdgcn_wmma_f32_16x16x32_bf16(false, fa1, false, fb,
                                                           (short)0, acc[1][tn], false, false);
    }
    __syncthreads();
  }

  // epilogue: C = acc + bias
#pragma unroll
  for (int tm = 0; tm < 2; ++tm)
#pragma unroll
    for (int tn = 0; tn < 4; ++tn) {
      int n = n0 + wn + tn * 16 + lan;
      float bv = bias[n];
#pragma unroll
      for (int v = 0; v < 8; ++v) {
        int m = m0 + wm + tm * 16 + v + half * 8;
        C[(size_t)m * N + n] = acc[tm][tn][v] + bv;
      }
    }
}

// ---------------------------------------------------------------------------
// Fused alpha kernel.
// Per block: 16 batches (b0..b0+15) x 128 output columns (n0..n0+127),
// all X=16 skip rows. GEMM rows r=0..255 map to (x=r>>4, bi=r&15).
// Accumulator tile (tm,tn) covers exactly one x = wm/16 + tm, bi = v+half*8.
//   alpha_raw[b,x,n] = sum_k skip_c[b,x,k] * aWhh[k,n]          (bf16 WMMA)
//   alpha = sigmoid(alpha_raw + alpha_wi[b,n])   (alpha_bias folded into awi)
//   softmax over {i} U {alpha masked by x<skip_count}, merge with {g, skip_c}
//   c1, h1 written to d_out.
// Waves: 4 along M (4 m-tiles each) x 2 along N (4 n-tiles each).
// ---------------------------------------------------------------------------
__global__ __launch_bounds__(256)
void alpha_fuse_kernel(const float* __restrict__ skip_c,
                       const float* __restrict__ aWhh,
                       const float* __restrict__ gates,  // [B,1536] raw (bias incl.)
                       const float* __restrict__ awi,    // [B,512] inp@aWih + abias
                       const float* __restrict__ c0,
                       const int*   __restrict__ scnt,
                       float* __restrict__ out) {        // h1 [B,H] then c1 [B,H]
  __shared__ unsigned short ldsA[256 * LDP];
  __shared__ unsigned short ldsB[128 * LDP];
  __shared__ float sden[16 * 128];
  __shared__ float snum[16 * 128];
  __shared__ float sawi[16 * 128];
  __shared__ int   sscnt[16];

  const int tid  = threadIdx.x;
  const int lane = tid & 31;
  const int wave = tid >> 5;
  const int half = lane >> 4;
  const int lan  = lane & 15;
  const int b0 = blockIdx.x * 16;
  const int n0 = blockIdx.y * 128;
  const int wm = (wave & 3) * 64;   // wave M offset (4 m-tiles)
  const int wn = (wave >> 2) * 64;  // wave N offset (4 n-tiles)

  v8f acc[4][4];
#pragma unroll
  for (int a = 0; a < 4; ++a)
#pragma unroll
    for (int b = 0; b < 4; ++b) acc[a][b] = v8f{};

  float2 rA[16];
  auto loadA = [&](int k0) {
#pragma unroll
    for (int e = 0; e < 16; ++e) {
      int i = tid + e * 256;
      int r = i >> 4, c2 = (i & 15) * 2;
      int x = r >> 4, bi = r & 15;
      rA[e] = *(const float2*)(skip_c + (size_t)((b0 + bi) * XX + x) * HH + k0 + c2);
    }
  };

  loadA(0);

  // ---- GEMM: M=256 (x,bi), N=128, K=512 ----
  for (int k0 = 0; k0 < 512; k0 += 32) {
#pragma unroll
    for (int e = 0; e < 16; ++e) {            // stage A 256x32 from regs
      int i = tid + e * 256;
      int r = i >> 4, c2 = (i & 15) * 2;
      *(unsigned*)(ldsA + r * LDP + c2) = pack2bf(rA[e].x, rA[e].y);
    }
#pragma unroll
    for (int e = 0; e < 8; ++e) {             // stage B^T 128x32 (L2-resident)
      int i = tid + e * 256;
      int k2 = (i >> 7) * 2, nc = i & 127;
      *(unsigned*)(ldsB + nc * LDP + k2) =
          pack2bf(aWhh[(size_t)(k0 + k2) * HH + n0 + nc],
                  aWhh[(size_t)(k0 + k2 + 1) * HH + n0 + nc]);
    }
    __syncthreads();

    if (k0 + 32 < 512) loadA(k0 + 32);        // pipeline skip_c stream

    v16bf fa[4];
#pragma unroll
    for (int tm = 0; tm < 4; ++tm)
      fa[tm] = load_frag_a(ldsA, wm + tm * 16 + lan, half);
#pragma unroll
    for (int tn = 0; tn < 4; ++tn) {
      v16bf fb = load_frag_b(ldsB, wn + tn * 16 + lan, half);
#pragma unroll
      for (int tm = 0; tm < 4; ++tm)
        acc[tm][tn] = __builtin_amdgcn_wmma_f32_16x16x32_bf16(false, fa[tm], false, fb,
                                                              (short)0, acc[tm][tn], false, false);
    }
    __syncthreads();
  }

  // ---- init reduction buffers with the i-gate term; hoist awi/scnt to LDS ----
  if (tid < 16) sscnt[tid] = scnt[b0 + tid];
#pragma unroll
  for (int e = 0; e < 8; ++e) {
    int idx = tid + e * 256;                  // 0..2047 = bi*128 + nl
    int bi = idx >> 7, nl = idx & 127;
    int b = b0 + bi, n = n0 + nl;
    float iraw = gates[(size_t)b * 1536 + n];
    float graw = gates[(size_t)b * 1536 + 1024 + n];
    float isig = 1.f / (1.f + __expf(-iraw));
    float g    = tanhf(graw);
    float ei   = __expf(isig);
    sden[idx] = ei;
    snum[idx] = ei * g;
    sawi[idx] = awi[(size_t)b * HH + n];
  }
  __syncthreads();

  // ---- alpha contributions: register-reduce over tm (4 x's), then one
  //      ds_add_f32 pair per (tn, v) ----
#pragma unroll
  for (int tn = 0; tn < 4; ++tn) {
    int nl = wn + tn * 16 + lan;
    int n  = n0 + nl;
#pragma unroll
    for (int v = 0; v < 8; ++v) {
      int bi = v + half * 8;
      int b  = b0 + bi;
      float aw = sawi[bi * 128 + nl];
      int   sc = sscnt[bi];
      float dsum = 0.f, nsum = 0.f;
#pragma unroll
      for (int tm = 0; tm < 4; ++tm) {
        int x = (wm >> 4) + tm;
        float a   = acc[tm][tn][v] + aw;
        float sig = 1.f / (1.f + __expf(-a));
        float ev  = (x < sc) ? __expf(sig) : 0.f;
        dsum += ev;
        nsum += ev * skip_c[(size_t)(b * XX + x) * HH + n];
      }
      atomicAdd(&sden[bi * 128 + nl], dsum);
      atomicAdd(&snum[bi * 128 + nl], nsum);
    }
  }
  __syncthreads();

  // ---- finalize c1, h1 ----
#pragma unroll
  for (int e = 0; e < 8; ++e) {
    int idx = tid + e * 256;
    int bi = idx >> 7, nl = idx & 127;
    int b = b0 + bi, n = n0 + nl;
    float iraw = gates[(size_t)b * 1536 + n];
    float oraw = gates[(size_t)b * 1536 + 512 + n];
    float graw = gates[(size_t)b * 1536 + 1024 + n];
    float isig = 1.f / (1.f + __expf(-iraw));
    float o    = 1.f / (1.f + __expf(-oraw));
    float g    = tanhf(graw);
    float c1   = snum[idx] / sden[idx];
    if (sscnt[bi] == 0) {
      c1 = (1.f - isig) * c0[(size_t)b * HH + n] + isig * g;
    }
    float h1 = o * tanhf(c1);
    out[(size_t)b * HH + n] = h1;
    out[(size_t)BB * HH + (size_t)b * HH + n] = c1;
  }
}

// ---------------------------------------------------------------------------
extern "C" void kernel_launch(void* const* d_in, const int* in_sizes, int n_in,
                              void* d_out, int out_size, void* d_ws, size_t ws_size,
                              hipStream_t stream) {
  const float* inp    = (const float*)d_in[0];
  const float* skip_c = (const float*)d_in[1];
  const float* h0     = (const float*)d_in[2];
  const float* c0     = (const float*)d_in[3];
  const float* Wih    = (const float*)d_in[4];
  const float* Whh    = (const float*)d_in[5];
  const float* aWih   = (const float*)d_in[6];
  const float* aWhh   = (const float*)d_in[7];
  const float* bias   = (const float*)d_in[8];
  const float* abias  = (const float*)d_in[9];
  const int*   scnt   = (const int*)d_in[10];
  float* out   = (float*)d_out;
  float* gates = (float*)d_ws;                       // [4096,1536]
  float* awi   = gates + (size_t)BB * 1536;          // [4096,512]

  dim3 blk(256);

  // gates = [inp|h0] @ [Wih;Whh] + bias   (M=4096, K=1024, N=1536)
  gemm_bf16_kernel<<<dim3(BB / 128, 1536 / 128), blk, 0, stream>>>(
      inp, h0, Wih, Whh, bias, gates, 1536, 1024, 512, 1536);

  // alpha_wi = inp @ aWih + abias          (M=4096, K=512, N=512)
  gemm_bf16_kernel<<<dim3(BB / 128, HH / 128), blk, 0, stream>>>(
      inp, inp, aWih, aWih, abias, awi, HH, 512, 512, HH);

  // fused alpha GEMM + softmax-merge + LSTM pointwise
  alpha_fuse_kernel<<<dim3(BB / 16, HH / 128), blk, 0, stream>>>(
      skip_c, aWhh, gates, awi, c0, scnt, out);
}
